// VQVAE_82463372083811
// MI455X (gfx1250) — matrix operand order repaired and checked by reading, not compiled, not run on hardware
//
#include <hip/hip_runtime.h>
#include <hip/hip_bf16.h>

// ---------- types ----------
typedef __bf16 v16bf __attribute__((ext_vector_type(16)));
typedef __bf16 v8bf  __attribute__((ext_vector_type(8)));
typedef float  v8f   __attribute__((ext_vector_type(8)));
typedef unsigned short u16_t;

#define BZ      16384
#define IN_DIM  1024
#define HID     400
#define HID_P   416      // 400 padded up to multiple of 32
#define EDIM    256
#define ENUM    2048

// ---------- helpers ----------
__device__ __forceinline__ u16_t f32_to_bf16(float f) {
    union { float f; unsigned int u; } v; v.f = f;
    unsigned int u = v.u;
    u += 0x7FFFu + ((u >> 16) & 1u);   // round-to-nearest-even
    return (u16_t)(u >> 16);
}

// A-fragment: 16 contiguous bf16 per lane, [p, p+16) halfwords
__device__ __forceinline__ v16bf load_fragA(const u16_t* p) {
    v8bf lo = *reinterpret_cast<const v8bf*>(p);
    v8bf hi = *reinterpret_cast<const v8bf*>(p + 8);
    v16bf r;
#pragma unroll
    for (int i = 0; i < 8; ++i) { r[i] = lo[i]; r[8 + i] = hi[i]; }
    return r;
}

// B-fragment: 16 contiguous bf16 per lane
__device__ __forceinline__ v16bf load_fragB(const u16_t* p) {
    v8bf lo = *reinterpret_cast<const v8bf*>(p);
    v8bf hi = *reinterpret_cast<const v8bf*>(p + 8);
    v16bf r;
#pragma unroll
    for (int i = 0; i < 8; ++i) { r[i] = lo[i]; r[8 + i] = hi[i]; }
    return r;
}

// ---------- f32 -> bf16 conversion with row and K padding ----------
__global__ void cvt_pad_kernel(const float* __restrict__ src, u16_t* __restrict__ dst,
                               int rows_in, int rows_out, int K, int Kp) {
    size_t total = (size_t)rows_out * Kp;
    for (size_t i = (size_t)blockIdx.x * blockDim.x + threadIdx.x; i < total;
         i += (size_t)gridDim.x * blockDim.x) {
        size_t r = i / Kp;
        int    k = (int)(i - r * Kp);
        dst[i] = (r < (size_t)rows_in && k < K)
                   ? f32_to_bf16(src[r * (size_t)K + k]) : (u16_t)0;
    }
}

// ---------- zero-padded f32 bias copy ----------
__global__ void pad_bias_kernel(const float* __restrict__ src, float* __restrict__ dst,
                                int n, int np) {
    int i = blockIdx.x * blockDim.x + threadIdx.x;
    if (i < np) dst[i] = (i < n) ? src[i] : 0.0f;
}

// ---------- ||emb_j||^2 ----------
__global__ void emb_norm_kernel(const float* __restrict__ emb, float* __restrict__ out) {
    __shared__ float s[256];
    int r = blockIdx.x;
    float v = emb[(size_t)r * EDIM + threadIdx.x];
    s[threadIdx.x] = v * v;
    __syncthreads();
    for (int off = 128; off > 0; off >>= 1) {
        if ((int)threadIdx.x < off) s[threadIdx.x] += s[threadIdx.x + off];
        __syncthreads();
    }
    if (threadIdx.x == 0) out[r] = s[0];
}

// ---------- generic WMMA GEMM: C = act(A * W^T + bias) ----------
// A: (M x K) bf16 row-major (lda), W: (N x K) bf16 row-major (ldb)
// Wave tile: 64 rows x 32 cols (4 M-reps x 2 N-reps -> 8 WMMAs per K-step)
// ACT: 0=none 1=relu 2=sigmoid ; OUT_BF16: store bf16 else f32
template <int ACT, bool OUT_BF16>
__global__ void gemm_wmma_kernel(const u16_t* __restrict__ A, int lda,
                                 const u16_t* __restrict__ W, int ldb,
                                 const float* __restrict__ bias,
                                 void* __restrict__ out, int ldo,
                                 int M, int N, int K) {
    const int lane = threadIdx.x & 31;
    const int wave = threadIdx.x >> 5;
    const int half = lane >> 4;       // 0: lanes 0-15, 1: lanes 16-31
    const int l15  = lane & 15;

    const int NT = N >> 5;            // 32-col tiles
    const int MT = M >> 6;            // 64-row tiles
    int t = blockIdx.x * 8 + wave;
    if (t >= MT * NT) return;
    int mt = t / NT, nt = t - mt * NT;
    int m0 = mt << 6, n0 = nt << 5;

    v8f acc[4][2];
#pragma unroll
    for (int i = 0; i < 4; ++i) {
        acc[i][0] = (v8f)0.0f;
        acc[i][1] = (v8f)0.0f;
    }

    const u16_t* brow0 = W + (size_t)(n0 + l15) * ldb + half * 16;
    const u16_t* brow1 = brow0 + (size_t)16 * ldb;
    const u16_t* arow0 = A + (size_t)(m0 + l15) * lda + half * 16;

    for (int k0 = 0; k0 < K; k0 += 32) {
        v16bf bf0 = load_fragB(brow0 + k0);
        v16bf bf1 = load_fragB(brow1 + k0);
#pragma unroll
        for (int i = 0; i < 4; ++i) {
            v16bf af = load_fragA(arow0 + (size_t)i * 16 * lda + k0);
            acc[i][0] = __builtin_amdgcn_wmma_f32_16x16x32_bf16(
                false, af, false, bf0, (short)0, acc[i][0], false, false);
            acc[i][1] = __builtin_amdgcn_wmma_f32_16x16x32_bf16(
                false, af, false, bf1, (short)0, acc[i][1], false, false);
        }
    }

    const float bv0 = bias[n0 + l15];
    const float bv1 = bias[n0 + 16 + l15];
#pragma unroll
    for (int i = 0; i < 4; ++i) {
        int rbase = m0 + i * 16 + half * 8;
#pragma unroll
        for (int j = 0; j < 2; ++j) {
            float bv = j ? bv1 : bv0;
            int   col = n0 + j * 16 + l15;
#pragma unroll
            for (int r = 0; r < 8; ++r) {
                float v = acc[i][j][r] + bv;
                if (ACT == 1) v = v > 0.0f ? v : 0.0f;
                if (ACT == 2) v = 1.0f / (1.0f + __expf(-v));
                size_t off = (size_t)(rbase + r) * ldo + col;
                if (OUT_BF16) ((u16_t*)out)[off] = f32_to_bf16(v);
                else          ((float*)out)[off] = v;
            }
        }
    }
}

// ---------- VQ: fused score GEMM + argmin ----------
// dist(i,j) = ||e_j||^2 - 2 * z_e[i].e_j   (||z_e||^2 is row-constant)
__global__ void vq_argmin_kernel(const u16_t* __restrict__ zeb,   // (BZ x 256) bf16
                                 const u16_t* __restrict__ embb,  // (2048 x 256) bf16
                                 const float* __restrict__ enorm, // (2048)
                                 int* __restrict__ outIdx) {
    const int lane = threadIdx.x & 31;
    const int wave = threadIdx.x >> 5;
    const int half = lane >> 4;
    const int l15  = lane & 15;
    const int m0   = (blockIdx.x * 8 + wave) * 16;

    // preload this wave's 16x256 z_e strip into registers (8 A fragments)
    const u16_t* arow = zeb + (size_t)(m0 + l15) * EDIM + half * 16;
    v16bf af[8];
#pragma unroll
    for (int kt = 0; kt < 8; ++kt) af[kt] = load_fragA(arow + kt * 32);

    float bv[8];
    int   bi[8];
#pragma unroll
    for (int r = 0; r < 8; ++r) { bv[r] = 3.4e38f; bi[r] = 0; }

    for (int ntile = 0; ntile < (ENUM >> 4); ++ntile) {
        const u16_t* brow = embb + (size_t)(ntile * 16 + l15) * EDIM + half * 16;
        v8f acc = (v8f)0.0f;
#pragma unroll
        for (int kt = 0; kt < 8; ++kt) {
            v16bf bf = load_fragB(brow + kt * 32);
            acc = __builtin_amdgcn_wmma_f32_16x16x32_bf16(
                false, af[kt], false, bf, (short)0, acc, false, false);
        }
        const int   n  = ntile * 16 + l15;   // column this lane holds
        const float en = enorm[n];
#pragma unroll
        for (int r = 0; r < 8; ++r) {
            float d = en - 2.0f * acc[r];
            if (d < bv[r]) { bv[r] = d; bi[r] = n; }
        }
    }

    // butterfly reduce across the 16 lanes of each half
#pragma unroll
    for (int r = 0; r < 8; ++r) {
        float v = bv[r]; int j = bi[r];
#pragma unroll
        for (int m = 1; m < 16; m <<= 1) {
            float ov = __shfl_xor(v, m, 32);
            int   oj = __shfl_xor(j, m, 32);
            if (ov < v) { v = ov; j = oj; }
        }
        if (l15 == 0) outIdx[m0 + half * 8 + r] = j;
    }
}

// ---------- gather z_q = emb[idx] (bf16) ----------
__global__ void gather_kernel(const int* __restrict__ idx,
                              const u16_t* __restrict__ embb,
                              u16_t* __restrict__ zqb) {
    int m = blockIdx.x;
    int j = idx[m];
    const unsigned int* s = (const unsigned int*)(embb + (size_t)j * EDIM);
    unsigned int*       d = (unsigned int*)(zqb + (size_t)m * EDIM);
    d[threadIdx.x] = s[threadIdx.x];   // 128 threads x 4B = 256 bf16
}

// ---------- host launcher ----------
extern "C" void kernel_launch(void* const* d_in, const int* in_sizes, int n_in,
                              void* d_out, int out_size, void* d_ws, size_t ws_size,
                              hipStream_t stream) {
    const float* x   = (const float*)d_in[0];
    const float* W1  = (const float*)d_in[1];
    const float* b1  = (const float*)d_in[2];
    const float* W2  = (const float*)d_in[3];
    const float* b2  = (const float*)d_in[4];
    const float* W3  = (const float*)d_in[5];
    const float* b3  = (const float*)d_in[6];
    const float* W4  = (const float*)d_in[7];
    const float* b4  = (const float*)d_in[8];
    const float* emb = (const float*)d_in[9];
    float* out = (float*)d_out;

    // workspace carve-out (256B aligned)
    size_t off = 0;
    auto carve = [&](size_t bytes) -> void* {
        off = (off + 255) & ~(size_t)255;
        void* p = (char*)d_ws + off;
        off += bytes;
        return p;
    };
    u16_t* xb    = (u16_t*)carve((size_t)BZ * IN_DIM * 2);
    u16_t* W1b   = (u16_t*)carve((size_t)HID_P * IN_DIM * 2);   // N-padded to 416 rows
    u16_t* W2b   = (u16_t*)carve((size_t)EDIM * HID_P * 2);     // K-padded to 416
    u16_t* W3b   = (u16_t*)carve((size_t)HID_P * EDIM * 2);     // N-padded to 416 rows
    u16_t* W4b   = (u16_t*)carve((size_t)IN_DIM * HID_P * 2);   // K-padded to 416
    u16_t* embb  = (u16_t*)carve((size_t)ENUM * EDIM * 2);
    float* en    = (float*)carve((size_t)ENUM * 4);
    float* b1p   = (float*)carve((size_t)HID_P * 4);
    float* b3p   = (float*)carve((size_t)HID_P * 4);
    u16_t* h1b   = (u16_t*)carve((size_t)BZ * HID_P * 2);
    u16_t* zeb   = (u16_t*)carve((size_t)BZ * EDIM * 2);
    int*   idx   = (int*)  carve((size_t)BZ * 4);
    u16_t* zqb   = (u16_t*)carve((size_t)BZ * EDIM * 2);
    u16_t* h3b   = (u16_t*)carve((size_t)BZ * HID_P * 2);
    (void)ws_size; (void)in_sizes; (void)n_in; (void)out_size;

    // conversions (pad rows and/or K with zeros)
    const int CT = 256;
    auto cvt = [&](const float* s, u16_t* d, int ri, int ro, int K, int Kp) {
        size_t total = (size_t)ro * Kp;
        int blocks = (int)((total + CT - 1) / CT);
        if (blocks > 4096) blocks = 4096;
        cvt_pad_kernel<<<blocks, CT, 0, stream>>>(s, d, ri, ro, K, Kp);
    };
    cvt(x,   xb,   BZ,     BZ,     IN_DIM, IN_DIM);
    cvt(W1,  W1b,  HID,    HID_P,  IN_DIM, IN_DIM);
    cvt(W2,  W2b,  EDIM,   EDIM,   HID,    HID_P);
    cvt(W3,  W3b,  HID,    HID_P,  EDIM,   EDIM);
    cvt(W4,  W4b,  IN_DIM, IN_DIM, HID,    HID_P);
    cvt(emb, embb, ENUM,   ENUM,   EDIM,   EDIM);
    pad_bias_kernel<<<(HID_P + 255) / 256, 256, 0, stream>>>(b1, b1p, HID, HID_P);
    pad_bias_kernel<<<(HID_P + 255) / 256, 256, 0, stream>>>(b3, b3p, HID, HID_P);
    emb_norm_kernel<<<ENUM, 256, 0, stream>>>(emb, en);

    auto gemm_grid = [](int M, int N) { return ((M >> 6) * (N >> 5) + 7) / 8; };

    // h1 = relu(x W1^T + b1)   (bf16 out, N padded to 416 -> pad cols written as 0)
    gemm_wmma_kernel<1, true><<<gemm_grid(BZ, HID_P), 256, 0, stream>>>(
        xb, IN_DIM, W1b, IN_DIM, b1p, h1b, HID_P, BZ, HID_P, IN_DIM);
    // z_e = h1 W2^T + b2       (bf16 out)
    gemm_wmma_kernel<0, true><<<gemm_grid(BZ, EDIM), 256, 0, stream>>>(
        h1b, HID_P, W2b, HID_P, b2, zeb, EDIM, BZ, EDIM, HID_P);
    // fused distance GEMM + argmin
    vq_argmin_kernel<<<BZ / 16 / 8, 256, 0, stream>>>(zeb, embb, en, idx);
    // z_q = emb[idx]
    gather_kernel<<<BZ, 128, 0, stream>>>(idx, embb, zqb);
    // h3 = relu(z_q W3^T + b3) (bf16 out, N padded to 416 -> pad cols written as 0)
    gemm_wmma_kernel<1, true><<<gemm_grid(BZ, HID_P), 256, 0, stream>>>(
        zqb, EDIM, W3b, EDIM, b3p, h3b, HID_P, BZ, HID_P, EDIM);
    // out = sigmoid(h3 W4^T + b4)  (f32 out)
    gemm_wmma_kernel<2, false><<<gemm_grid(BZ, IN_DIM), 256, 0, stream>>>(
        h3b, HID_P, W4b, HID_P, b4, out, IN_DIM, BZ, IN_DIM, HID_P);
}
